// RNN_54022098649219
// MI455X (gfx1250) — matrix operand order repaired and checked by reading, not compiled, not run on hardware
//
#include <hip/hip_runtime.h>

#define T_STEPS 2048
#define HID 256
#define BATCH 256
#define POUT 24
#define LDH 264      // padded ushort row stride for s_h: 4-bank shift per row -> conflict-free
#define SXL 65       // padded float row stride for staged x
#define CHUNK 64     // timesteps of x staged in LDS at a time

typedef __attribute__((ext_vector_type(16))) __bf16 v16bf;
typedef __attribute__((ext_vector_type(8)))  float  v8f;

union Frag {
    v16bf          v;
    unsigned short s[16];
    uint4          q[2];
};

static __device__ __forceinline__ unsigned short f2bf(float f) {
    unsigned int u = __builtin_bit_cast(unsigned int, f);
    unsigned int r = u + 0x7FFFu + ((u >> 16) & 1u);   // round-to-nearest-even
    return (unsigned short)(r >> 16);
}
static __device__ __forceinline__ float bf2f(unsigned short s) {
    unsigned int u = ((unsigned int)s) << 16;
    return __builtin_bit_cast(float, u);
}

// Branch-free tanh via hardware transcendentals:
//   tanh(x) = 1 - 2/(exp2(x * 2*log2(e)) + 1)
// v_exp_f32 + v_rcp_f32, no EXEC divergence, saturates correctly at +/-inf.
static __device__ __forceinline__ float fast_tanh(float x) {
    float e = __builtin_amdgcn_exp2f(x * 2.885390081777927f);   // e^(2x)
    float r = __builtin_amdgcn_rcpf(e + 1.0f);
    return __builtin_fmaf(-2.0f, r, 1.0f);
}

// One workgroup per 16-row batch tile; 16 waves; wave w owns hidden columns
// [16w, 16w+16). Persistent over all T timesteps; h lives in LDS as bf16,
// W_hh B-fragments live in registers (time-invariant).
__global__ __launch_bounds__(512, 1)
void rnn_scan_wmma(const float* __restrict__ x,      // [B,T]
                   const float* __restrict__ W_ih,   // [H,1]
                   const float* __restrict__ W_hh,   // [H,H]
                   const float* __restrict__ b_ih,   // [H]
                   const float* __restrict__ b_hh,   // [H]
                   const float* __restrict__ W_out,  // [P,H]
                   const float* __restrict__ b_out,  // [P]
                   float* __restrict__ out)          // [B,P]
{
    __shared__ __align__(16) unsigned short s_h[16 * LDH];
    __shared__ float s_x[16 * SXL];

    const int tid  = threadIdx.x;
    const int lane = tid & 31;
    const int nt   = tid >> 5;     // wave id == N-tile index (0..15)
    const int hi   = lane >> 4;    // half-wave selector
    const int n    = lane & 15;
    const int b0   = blockIdx.x * 16;

    // h(0) = 0
    for (int e = tid; e < 16 * LDH; e += 512) s_h[e] = 0;

    // Per-lane input-projection constants for this lane's hidden unit.
    const int   hidx = nt * 16 + n;
    const float wih  = W_ih[hidx];
    const float bias = b_ih[hidx] + b_hh[hidx];

    // Preload B-fragments: M[b,j] = sum_k h[b,k]*W_hh[j,k]
    //   => B[k][n] = W_hh[col][k], col = nt*16+n.
    // B k order used here (same as A): lanes 0-15 hold K=kb*32+0..15, lanes 16-31 K=kb*32+16..31.
    Frag bfr[8];
    {
        const int col = nt * 16 + n;
        #pragma unroll
        for (int kb = 0; kb < 8; ++kb) {
            const float4* p = (const float4*)(W_hh + col * HID + kb * 32 + hi * 16);
            #pragma unroll
            for (int q4 = 0; q4 < 4; ++q4) {
                float4 w = p[q4];
                bfr[kb].s[q4 * 4 + 0] = f2bf(w.x);
                bfr[kb].s[q4 * 4 + 1] = f2bf(w.y);
                bfr[kb].s[q4 * 4 + 2] = f2bf(w.z);
                bfr[kb].s[q4 * 4 + 3] = f2bf(w.w);
            }
        }
    }
    __syncthreads();

    for (int t = 0; t < T_STEPS; ++t) {
        if ((t & (CHUNK - 1)) == 0) {
            // Stage x[b0+r, t..t+CHUNK) -> LDS; 512 threads x 2 floats.
            const int r  = tid >> 5;
            const int c2 = lane * 2;
            const float2 v =
                *(const float2*)(x + (size_t)(b0 + r) * T_STEPS + t + c2);
            s_x[r * SXL + c2]     = v.x;
            s_x[r * SXL + c2 + 1] = v.y;
            if (t + CHUNK < T_STEPS)
                __builtin_prefetch(
                    x + (size_t)(b0 + r) * T_STEPS + t + CHUNK + c2, 0, 3);
            __syncthreads();
        }

        // 16x16 output tile per wave: A = h_tile (bf16 LDS), B in registers.
        // Two accumulator chains (even/odd K-blocks) -> dependent XDL chain of
        // 4 instead of 8; merged with one vector add at the end.
        v8f c0 = {};
        v8f c1 = {};
        #pragma unroll
        for (int kb = 0; kb < 8; kb += 2) {
            Frag a0, a1;
            const int m = lane & 15;   // A-matrix row = lane (mod 16)
            const uint4* pa0 =
                (const uint4*)(s_h + m * LDH + (kb + 0) * 32 + hi * 16);
            const uint4* pa1 =
                (const uint4*)(s_h + m * LDH + (kb + 1) * 32 + hi * 16);
            a0.q[0] = pa0[0];          // K = kb*32 + hi*16 + 0..7
            a0.q[1] = pa0[1];          // K = kb*32 + hi*16 + 8..15
            a1.q[0] = pa1[0];
            a1.q[1] = pa1[1];
            c0 = __builtin_amdgcn_wmma_f32_16x16x32_bf16(
                     false, a0.v, false, bfr[kb + 0].v, (short)0, c0, false, false);
            c1 = __builtin_amdgcn_wmma_f32_16x16x32_bf16(
                     false, a1.v, false, bfr[kb + 1].v, (short)0, c1, false, false);
        }
        v8f c = c0 + c1;

        __syncthreads();   // all waves finished reading s_h

        // h_new = tanh(matmul + x*W_ih + biases); write bf16 back to LDS.
        const int tc = t & (CHUNK - 1);
        unsigned short hn[8];
        #pragma unroll
        for (int i = 0; i < 8; ++i) {
            const int m = i + hi * 8;                     // C/D row mapping
            const float pre = __builtin_fmaf(s_x[m * SXL + tc], wih, bias);
            hn[i] = f2bf(fast_tanh(c[i] + pre));
        }
        #pragma unroll
        for (int i = 0; i < 8; ++i) {
            const int m = i + hi * 8;
            s_h[m * LDH + hidx] = hn[i];
        }

        __syncthreads();   // new h visible to all waves
    }

    // Head: out[b0+r, p] = b_out[p] + sum_k h[r,k] * W_out[p,k]
    if (tid < 16 * POUT) {
        const int r = tid / POUT;
        const int p = tid % POUT;
        float acc = b_out[p];
        for (int k = 0; k < HID; ++k)
            acc += bf2f(s_h[r * LDH + k]) * W_out[p * HID + k];
        out[(size_t)(b0 + r) * POUT + p] = acc;
    }
}

extern "C" void kernel_launch(void* const* d_in, const int* in_sizes, int n_in,
                              void* d_out, int out_size, void* d_ws, size_t ws_size,
                              hipStream_t stream) {
    const float* x     = (const float*)d_in[0];
    const float* W_ih  = (const float*)d_in[1];
    const float* W_hh  = (const float*)d_in[2];
    const float* b_ih  = (const float*)d_in[3];
    const float* b_hh  = (const float*)d_in[4];
    const float* W_out = (const float*)d_in[5];
    const float* b_out = (const float*)d_in[6];
    float* out = (float*)d_out;

    rnn_scan_wmma<<<dim3(BATCH / 16), dim3(512), 0, stream>>>(
        x, W_ih, W_hh, b_ih, b_hh, W_out, b_out, out);
}
